// CausalSelfAttention_55903294325242
// MI455X (gfx1250) — matrix operand order repaired and checked by reading, not compiled, not run on hardware
//
#include <hip/hip_runtime.h>
#include <hip/hip_bf16.h>
#include <stdint.h>

typedef __attribute__((ext_vector_type(16))) __bf16 v16bf;
typedef __attribute__((ext_vector_type(8)))  float  v8f;

#define B_  4
#define T_  2048
#define DM  1024
#define NH  16
#define HD  64
#define BT  (B_*T_)   // 8192

__device__ __forceinline__ unsigned short f2bf(float f){
  unsigned int x = __float_as_uint(f);
  x += 0x7FFFu + ((x >> 16) & 1u);          // round-to-nearest-even
  return (unsigned short)(x >> 16);
}
__device__ __forceinline__ float bf2f(unsigned short h){
  return __uint_as_float(((unsigned int)h) << 16);
}
// A-matrix (16x32 bf16) K index for VGPR j, lane-half h  (contiguous order k = 16h + 2j, matched by kloB)
__device__ __forceinline__ int kloA(int j, int h){
  return 2*j + 16*h;
}
// B-matrix (32x16 bf16) K index for VGPR j, lane-half h
__device__ __forceinline__ int kloB(int j, int h){
  return 2*j + 16*h;
}

union FragAB { v16bf v; unsigned int u[8]; };

// ---- CDNA5 async global->LDS copy (16B per lane), ASYNCcnt tracked ----
// Generic LDS pointer -> LDS offset is the low 32 bits (ISA 10.2 aperture:
// low 32 bits of the address); global address goes in a VGPR pair.
__device__ __forceinline__ void async_g2l_b128(void* lds, const void* g){
  unsigned loff = (unsigned)(uintptr_t)lds;
  unsigned long long ga = (unsigned long long)(uintptr_t)g;
  asm volatile("global_load_async_to_lds_b128 %0, %1, off"
               :: "v"(loff), "v"(ga) : "memory");
}
__device__ __forceinline__ void wait_async0(){
#if __has_builtin(__builtin_amdgcn_s_wait_asynccnt)
  __builtin_amdgcn_s_wait_asynccnt(0);
#else
  asm volatile("s_wait_asynccnt 0" ::: "memory");
#endif
}

// ---------------------------------------------------------------- cast f32->bf16
__global__ void cast_f32_bf16(const float* __restrict__ src,
                              unsigned short* __restrict__ dst, int n4){
  int i = blockIdx.x * blockDim.x + threadIdx.x;
  if (i >= n4) return;
  float4 f = ((const float4*)src)[i];
  uint2 o;
  o.x = (unsigned)f2bf(f.x) | ((unsigned)f2bf(f.y) << 16);
  o.y = (unsigned)f2bf(f.z) | ((unsigned)f2bf(f.w) << 16);
  ((uint2*)dst)[i] = o;
}

// ---------------------------------------------------------------- WMMA GEMM
// C(MxN,f32) = A(MxK,bf16) * B(KxN,bf16).  Block tile 128x64, 8 waves of 32x32.
// A tile staged via GLOBAL_LOAD_ASYNC_TO_LDS_B128; B tile K-pair packed manually.
// MODE 0: plain f32 store.  MODE 1: scatter bf16 into q/k/v^T buffers.
#define BMT 128
#define BNT 64
#define LDA 40      // elems/row: 16B-aligned rows for async b128 (80B = 20 dwords)
#define LDBP 68     // dwords/row of sBp (half-lane offset 8*68 % 64 == 32: disjoint banks)

template<int MODE>
__global__ __launch_bounds__(256) void gemm_bf16(
    const unsigned short* __restrict__ Ag, const unsigned short* __restrict__ Bg,
    int M, int N, int K,
    float* __restrict__ outf,
    unsigned short* __restrict__ qdst, unsigned short* __restrict__ kdst,
    unsigned short* __restrict__ vtdst)
{
  __shared__ unsigned short sA[BMT * LDA];
  __shared__ unsigned int   sBp[16 * LDBP];   // K-pairs x N, packed {k,k+1} bf16 per dword

  const int tid  = threadIdx.x;
  const int wid  = tid >> 5;
  const int lane = tid & 31;
  const int half = lane >> 4;
  const int ml   = lane & 15;
  const int wm   = wid & 3;     // wave row 0..3  (32 rows each)
  const int wn   = wid >> 2;    // wave col 0..1  (32 cols each)
  const int bm   = blockIdx.y * BMT;
  const int bn   = blockIdx.x * BNT;

  const int ar  = tid >> 1;            // 0..127  A-loader row
  const int ac  = (tid & 1) * 16;      // 0/16    A-loader col half
  const int bkk = tid >> 4;            // 0..15   B-loader K-pair
  const int bnb = (tid & 15) * 4;      // 0..60   B-loader col base

  v8f acc[2][2];
  #pragma unroll
  for (int i = 0; i < 2; i++)
    #pragma unroll
    for (int j = 0; j < 2; j++) acc[i][j] = (v8f)(0.0f);

  for (int kt = 0; kt < K; kt += 32) {
    // A tile 128x32 -> LDS via async DMA-style copy (2 x b128 per lane)
    {
      const unsigned short* ga = Ag + (size_t)(bm + ar) * K + kt + ac;
      unsigned short* la = &sA[ar * LDA + ac];
      async_g2l_b128(la,     ga);
      async_g2l_b128(la + 8, ga + 8);
    }
    // B tile 32x64 -> LDS, K-pair packed through VGPRs
    {
      const unsigned short* r0 = Bg + (size_t)(kt + 2*bkk) * N + bn + bnb;
      const unsigned short* r1 = r0 + N;
      uint2 lo = *(const uint2*)r0;
      uint2 hi = *(const uint2*)r1;
      const unsigned short* lp = (const unsigned short*)&lo;
      const unsigned short* hp = (const unsigned short*)&hi;
      uint4 pk;
      pk.x = (unsigned)lp[0] | ((unsigned)hp[0] << 16);
      pk.y = (unsigned)lp[1] | ((unsigned)hp[1] << 16);
      pk.z = (unsigned)lp[2] | ((unsigned)hp[2] << 16);
      pk.w = (unsigned)lp[3] | ((unsigned)hp[3] << 16);
      *(uint4*)&sBp[bkk * LDBP + bnb] = pk;
    }
    // prefetch next K-tile into L2 while this one computes
    if (kt + 32 < K) {
      __builtin_prefetch(Ag + (size_t)(bm + ar) * K + kt + 32 + ac, 0, 0);
      __builtin_prefetch(Bg + (size_t)(kt + 32 + 2*bkk) * N + bn + bnb, 0, 0);
    }
    wait_async0();
    __syncthreads();

    FragAB af[2], bfr[2];
    #pragma unroll
    for (int fm = 0; fm < 2; fm++) {
      int row = wm*32 + fm*16 + ml;
      #pragma unroll
      for (int j = 0; j < 8; j++)
        af[fm].u[j] = *(const unsigned int*)&sA[row * LDA + kloA(j, half)];
    }
    #pragma unroll
    for (int fn = 0; fn < 2; fn++) {
      int col = wn*32 + fn*16 + ml;
      #pragma unroll
      for (int j = 0; j < 8; j++)
        bfr[fn].u[j] = sBp[(j + 8*half) * LDBP + col];
    }
    #pragma unroll
    for (int fm = 0; fm < 2; fm++)
      #pragma unroll
      for (int fn = 0; fn < 2; fn++)
        acc[fm][fn] = __builtin_amdgcn_wmma_f32_16x16x32_bf16(
            false, af[fm].v, false, bfr[fn].v, (short)0, acc[fm][fn], false, false);
    __syncthreads();
  }

  #pragma unroll
  for (int fm = 0; fm < 2; fm++) {
    #pragma unroll
    for (int fn = 0; fn < 2; fn++) {
      #pragma unroll
      for (int r = 0; r < 8; r++) {
        int m = bm + wm*32 + fm*16 + r + 8*half;   // C/D layout: M = r + 8*half
        int n = bn + wn*32 + fn*16 + ml;           // N = lane%16
        float v = acc[fm][fn][r];
        if (MODE == 0) {
          outf[(size_t)m * N + n] = v;
        } else {
          int which = n >> 10;         // 0=q 1=k 2=v
          int n2 = n & 1023;
          int h  = n2 >> 6;
          int d  = n2 & 63;
          int b  = m >> 11;
          int t  = m & (T_ - 1);
          unsigned short hv = f2bf(v);
          size_t bh = (size_t)(b * NH + h);
          if      (which == 0) qdst [(bh * T_ + t) * HD + d] = hv;
          else if (which == 1) kdst [(bh * T_ + t) * HD + d] = hv;
          else                 vtdst[(bh * HD + d) * T_ + t] = hv;  // V stored transposed
        }
      }
    }
  }
}

// ---------------------------------------------------------------- RoPE (interleaved pairs)
__global__ void rope_kernel(unsigned short* __restrict__ buf,
                            const float* __restrict__ cosb,
                            const float* __restrict__ sinb, int npairs){
  int i = blockIdx.x * blockDim.x + threadIdx.x;
  if (i >= npairs) return;
  int d2 = i & 31;
  int t  = (i >> 5) & (T_ - 1);
  size_t base = (size_t)(i >> 5) * HD + 2*d2;   // (bh*T + t)*64 + 2*d2
  float x1 = bf2f(buf[base]);
  float x2 = bf2f(buf[base + 1]);
  float c = cosb[t * 32 + d2];
  float s = sinb[t * 32 + d2];
  buf[base]     = f2bf(x1 * c - x2 * s);
  buf[base + 1] = f2bf(x2 * c + x1 * s);
}

// ---------------------------------------------------------------- causal flash attention
// grid (T/256, B*H); 8 waves/block; each wave owns 32 query rows (2 M-frags) so
// every K/V fragment feeds 2 WMMAs.  q0 is 32-aligned -> only the diagonal
// KV tile needs the causal mask; all earlier tiles take the unmasked fast path.
// Scale/mask/softmax operate IN PLACE on the S accumulators to stay under 256 VGPRs.
__global__ __launch_bounds__(256) void attn_kernel(
    const unsigned short* __restrict__ qb, const unsigned short* __restrict__ kb,
    const unsigned short* __restrict__ vtb, unsigned short* __restrict__ aob)
{
  __shared__ unsigned short sP[8 * 1024];   // per-wave 32x32 bf16 transpose scratch
  const int tid  = threadIdx.x;
  const int wid  = tid >> 5;
  const int lane = tid & 31;
  const int half = lane >> 4;
  const int ml   = lane & 15;
  const int bh   = blockIdx.y;
  const int b    = bh >> 4;
  const int h    = bh & 15;
  const int q0   = blockIdx.x * 256 + wid * 32;   // multiple of 32

  const unsigned short* Q  = qb  + (size_t)bh * T_ * HD;
  const unsigned short* Kp = kb  + (size_t)bh * T_ * HD;
  const unsigned short* Vt = vtb + (size_t)bh * HD * T_;
  unsigned short* sPw = &sP[wid * 1024];

  FragAB aq[2][2];                              // [m-frag][k-step], resident
  #pragma unroll
  for (int mf = 0; mf < 2; mf++)
    #pragma unroll
    for (int i = 0; i < 2; i++)
      #pragma unroll
      for (int j = 0; j < 8; j++)
        aq[mf][i].u[j] = *(const unsigned int*)
            (Q + (size_t)(q0 + mf*16 + ml) * HD + i*32 + kloA(j, half));

  v8f o[2][4];
  #pragma unroll
  for (int mf = 0; mf < 2; mf++)
    #pragma unroll
    for (int fn = 0; fn < 4; fn++) o[mf][fn] = (v8f)(0.0f);
  float rm[2][8], rl[2][8];
  #pragma unroll
  for (int mf = 0; mf < 2; mf++)
    #pragma unroll
    for (int r = 0; r < 8; r++) { rm[mf][r] = -3.0e38f; rl[mf][r] = 0.0f; }

  auto tile = [&](int kv0, bool masked) {
    // ---- S = Q K^T : 4 B-frag loads, 8 WMMA
    v8f s[2][2];
    #pragma unroll
    for (int g = 0; g < 2; g++) {
      int keyb = kv0 + g*16;
      FragAB bk[2];
      #pragma unroll
      for (int i = 0; i < 2; i++)
        #pragma unroll
        for (int j = 0; j < 8; j++)
          bk[i].u[j] = *(const unsigned int*)
              (Kp + (size_t)(keyb + ml) * HD + i*32 + kloB(j, half));
      #pragma unroll
      for (int mf = 0; mf < 2; mf++) {
        v8f acc = (v8f)(0.0f);
        acc = __builtin_amdgcn_wmma_f32_16x16x32_bf16(false, aq[mf][0].v, false, bk[0].v,
                                                      (short)0, acc, false, false);
        acc = __builtin_amdgcn_wmma_f32_16x16x32_bf16(false, aq[mf][1].v, false, bk[1].v,
                                                      (short)0, acc, false, false);
        s[mf][g] = acc;
      }
    }
    // ---- scale (+ causal mask only on the diagonal tile) IN PLACE
    #pragma unroll
    for (int mf = 0; mf < 2; mf++)
      #pragma unroll
      for (int g = 0; g < 2; g++) {
        int key = kv0 + g*16 + ml;
        #pragma unroll
        for (int r = 0; r < 8; r++) {
          float vv = s[mf][g][r] * 0.125f;          // 1/sqrt(64)
          if (masked) {
            int qrow = q0 + mf*16 + r + 8*half;
            vv = (key > qrow) ? -3.0e38f : vv;
          }
          s[mf][g][r] = vv;
        }
      }
    // ---- online softmax (16-lane butterfly stays inside lane-half);
    //      rm becomes the new max immediately, only alpha stays live.
    float alpha[2][8];
    #pragma unroll
    for (int mf = 0; mf < 2; mf++)
      #pragma unroll
      for (int r = 0; r < 8; r++) {
        float v = fmaxf(s[mf][0][r], s[mf][1][r]);
        #pragma unroll
        for (int m = 1; m <= 8; m <<= 1)
          v = fmaxf(v, __shfl_xor(v, m, 32));
        float mnew = fmaxf(rm[mf][r], v);
        alpha[mf][r] = __expf(rm[mf][r] - mnew);
        rm[mf][r]    = mnew;
      }
    #pragma unroll
    for (int mf = 0; mf < 2; mf++) {
      float psum[8];
      #pragma unroll
      for (int r = 0; r < 8; r++) psum[r] = 0.0f;
      #pragma unroll
      for (int g = 0; g < 2; g++)
        #pragma unroll
        for (int r = 0; r < 8; r++) {
          float p = __expf(s[mf][g][r] - rm[mf][r]);
          psum[r] += p;
          sPw[(mf*16 + r + 8*half) * 32 + g*16 + ml] = f2bf(p);  // C-layout -> [m][k]
        }
      #pragma unroll
      for (int r = 0; r < 8; r++) {
        float v = psum[r];
        #pragma unroll
        for (int m = 1; m <= 8; m <<= 1)
          v += __shfl_xor(v, m, 32);
        rl[mf][r] = rl[mf][r] * alpha[mf][r] + v;
      }
      #pragma unroll
      for (int fn = 0; fn < 4; fn++)
        #pragma unroll
        for (int r = 0; r < 8; r++)
          o[mf][fn][r] *= alpha[mf][r];
    }
    // ---- read P back in A-fragment layout (wave-private LDS, DS ops in-order)
    asm volatile("s_wait_dscnt 0" ::: "memory");
    FragAB ap[2];
    #pragma unroll
    for (int mf = 0; mf < 2; mf++)
      #pragma unroll
      for (int j = 0; j < 8; j++)
        ap[mf].u[j] = *(const unsigned int*)&sPw[(mf*16 + ml) * 32 + kloA(j, half)];
    // ---- O += P V : 4 V-frag loads, 8 WMMA (V^T makes K-pairs contiguous)
    #pragma unroll
    for (int fn = 0; fn < 4; fn++) {
      FragAB bv;
      #pragma unroll
      for (int j = 0; j < 8; j++)
        bv.u[j] = *(const unsigned int*)
            (Vt + (size_t)(fn*16 + ml) * T_ + kv0 + kloB(j, half));
      #pragma unroll
      for (int mf = 0; mf < 2; mf++)
        o[mf][fn] = __builtin_amdgcn_wmma_f32_16x16x32_bf16(false, ap[mf].v, false, bv.v,
                                                            (short)0, o[mf][fn], false, false);
    }
  };

  for (int kv0 = 0; kv0 < q0; kv0 += 32) tile(kv0, false);  // unmasked fast path
  tile(q0, true);                                           // diagonal tile

  // ---- normalize + store bf16 into [B,T,D] for the projection GEMM
  #pragma unroll
  for (int mf = 0; mf < 2; mf++)
    #pragma unroll
    for (int fn = 0; fn < 4; fn++)
      #pragma unroll
      for (int r = 0; r < 8; r++) {
        int tq = q0 + mf*16 + r + 8*half;
        int dc = h * HD + fn*16 + ml;
        aob[((size_t)b * T_ + tq) * DM + dc] = f2bf(o[mf][fn][r] / rl[mf][r]);
      }
}

// ---------------------------------------------------------------- launch
extern "C" void kernel_launch(void* const* d_in, const int* in_sizes, int n_in,
                              void* d_out, int out_size, void* d_ws, size_t ws_size,
                              hipStream_t stream)
{
  const float* x     = (const float*)d_in[0];
  const float* cosb  = (const float*)d_in[1];
  const float* sinb  = (const float*)d_in[2];
  const float* wqkv  = (const float*)d_in[3];
  const float* wproj = (const float*)d_in[4];
  float* out = (float*)d_out;

  char* ws = (char*)d_ws;
  unsigned short* xb     = (unsigned short*)(ws);              // 16 MB
  unsigned short* wqkvb  = (unsigned short*)(ws + 16777216);   //  6 MB
  unsigned short* wprojb = (unsigned short*)(ws + 23068672);   //  2 MB
  unsigned short* qb     = (unsigned short*)(ws + 25165824);   // 16 MB [B,H,T,64]
  unsigned short* kb     = (unsigned short*)(ws + 41943040);   // 16 MB [B,H,T,64]
  unsigned short* vtb    = (unsigned short*)(ws + 58720256);   // 16 MB [B,H,64,T]
  unsigned short* aob    = (unsigned short*)(ws + 75497472);   // 16 MB [B,T,D]

  cast_f32_bf16<<<(8388608/4 + 255)/256, 256, 0, stream>>>(x, xb, 8388608/4);
  cast_f32_bf16<<<(3145728/4 + 255)/256, 256, 0, stream>>>(wqkv, wqkvb, 3145728/4);
  cast_f32_bf16<<<(1048576/4 + 255)/256, 256, 0, stream>>>(wproj, wprojb, 1048576/4);

  dim3 g1(3*DM/BNT, BT/BMT);   // (48, 64)
  gemm_bf16<1><<<g1, 256, 0, stream>>>(xb, wqkvb, BT, 3*DM, DM, nullptr, qb, kb, vtb);

  int npairs = B_ * NH * T_ * 32;
  rope_kernel<<<(npairs + 255)/256, 256, 0, stream>>>(qb, cosb, sinb, npairs);
  rope_kernel<<<(npairs + 255)/256, 256, 0, stream>>>(kb, cosb, sinb, npairs);

  dim3 g2(T_/256, B_*NH);      // (8, 64)
  attn_kernel<<<g2, 256, 0, stream>>>(qb, kb, vtb, aob);

  dim3 g3(DM/BNT, BT/BMT);     // (16, 64)
  gemm_bf16<0><<<g3, 256, 0, stream>>>(aob, wprojb, BT, DM, DM, out,
                                       nullptr, nullptr, nullptr);
}
